// BGEEmbeddingHeads_39659728011420
// MI455X (gfx1250) — matrix operand order repaired and checked by reading, not compiled, not run on hardware
//
#include <hip/hip_runtime.h>
#include <cstdint>

#define B_   32
#define S_   512
#define H_   1024
#define V_   250002
#define EPSN 1e-12f
#define APAD 8

typedef __bf16        bf16_t;
typedef __bf16        v16bf __attribute__((ext_vector_type(16)));
typedef float         v8f   __attribute__((ext_vector_type(8)));
typedef float         f4    __attribute__((ext_vector_type(4)));
typedef unsigned int  u4    __attribute__((ext_vector_type(4)));
typedef unsigned int  u2    __attribute__((ext_vector_type(2)));

union Frag16 { v16bf v; u4 q[2]; };
union Pack4  { bf16_t b[4]; u2 u; };

static __device__ __forceinline__ void splitBf16(float x, bf16_t& hi, bf16_t& lo) {
    hi = (bf16_t)x;
    lo = (bf16_t)(x - (float)hi);
}

// ---------------------------------------------------------------------------
// Kernel 0: one-time split of mv_w (f32) into bf16 hi/lo planes, stored in
// WMMA-B-fragment-swizzled order so GEMM loads are fully coalesced:
//   dst = ((nt*32 + c)*32 + lane)*16 + ke
// where nt = n/16 (N tile), c = k/32 (K chunk), lane = (k%32)/16*16 + n%16,
// ke = k%16.  Per (tile, chunk) a wave's 32 lanes occupy one contiguous 1KB per plane.
// ---------------------------------------------------------------------------
__global__ __launch_bounds__(256)
void convert_w_kernel(const float* __restrict__ mv_w,
                      bf16_t* __restrict__ whi,
                      bf16_t* __restrict__ wlo)
{
    const size_t i = ((size_t)blockIdx.x * 256 + threadIdx.x) * 4;
    const int n = (int)(i >> 10);
    const int k = (int)(i & 1023);
    const int nt = n >> 4, col = n & 15;
    const int c  = k >> 5, k32 = k & 31;
    const int lane = (k32 >> 4) * 16 + col;
    const int ke   = k32 & 15;
    const size_t dst = ((size_t)(nt * 32 + c) * 32 + lane) * 16 + ke;

    const f4 v = *reinterpret_cast<const f4*>(mv_w + i);
    Pack4 h, l;
    splitBf16(v.x, h.b[0], l.b[0]);
    splitBf16(v.y, h.b[1], l.b[1]);
    splitBf16(v.z, h.b[2], l.b[2]);
    splitBf16(v.w, h.b[3], l.b[3]);
    *reinterpret_cast<u2*>(whi + dst) = h.u;
    *reinterpret_cast<u2*>(wlo + dst) = l.u;
}

// ---------------------------------------------------------------------------
// Kernel 1: mv head GEMM (bf16x3 WMMA) fused with masked pooling + sparse head
// grid = 1024 blocks (16 tokens each), 256 threads = 8 wave32s
// ---------------------------------------------------------------------------
__global__ __launch_bounds__(256)
void mv_sparse_pool_kernel(const float*  __restrict__ hidden,
                           const int*    __restrict__ input_ids,
                           const float*  __restrict__ mask,
                           const float*  __restrict__ sparse_w,
                           const float*  __restrict__ sparse_b,
                           const bf16_t* __restrict__ whi,
                           const bf16_t* __restrict__ wlo,
                           const float*  __restrict__ mv_b,
                           float* __restrict__ out_sparse,
                           float* __restrict__ out_mv,
                           float* __restrict__ ws_pool)
{
    __shared__ __align__(16) bf16_t Ahi[16][512 + APAD];
    __shared__ __align__(16) bf16_t Alo[16][512 + APAD];
    __shared__ float swp[8][8];
    __shared__ float rowsq[16];

    const int t    = threadIdx.x;
    const int wave = t >> 5;
    const int lane = t & 31;
    const int l15  = lane & 15;         // A row / B,C column within tile
    const int hi16 = lane >> 4;         // half-wave selector
    const int hi8  = hi16 * 8;          // C-matrix row offset for lanes 16..31
    const int t0   = blockIdx.x * 16;   // first flat token index (b*S + s)
    const int b    = t0 / S_;
    const int s0   = t0 - b * S_;

    if (t < 16) rowsq[t] = 0.f;

    v8f acc[8];
#pragma unroll
    for (int j = 0; j < 8; ++j)
#pragma unroll
        for (int e = 0; e < 8; ++e) acc[j][e] = 0.f;

    const int par   = t >> 7;        // token parity handled by this thread
    const int hcolq = t & 127;       // float4 column within K-half
    float pA[8];
#pragma unroll
    for (int i = 0; i < 8; ++i) pA[i] = 0.f;

    for (int kh = 0; kh < 2; ++kh) {              // two K halves of 512
        // ---- stage 16x512 fp32 -> bf16 hi/lo planes in LDS; fuse pooling + sparse dot
        const f4 sw4 = *reinterpret_cast<const f4*>(sparse_w + kh * 512 + hcolq * 4);
        f4 pool; pool.x = pool.y = pool.z = pool.w = 0.f;
#pragma unroll
        for (int i = 0; i < 8; ++i) {
            const int tok = 2 * i + par;
            const f4 v = *reinterpret_cast<const f4*>(
                hidden + (size_t)(t0 + tok) * H_ + kh * 512 + hcolq * 4);
            const float m = mask[(size_t)b * S_ + s0 + tok];
            pool.x += m * v.x; pool.y += m * v.y; pool.z += m * v.z; pool.w += m * v.w;
            pA[i]  += v.x * sw4.x + v.y * sw4.y + v.z * sw4.z + v.w * sw4.w;
            Pack4 h, l;
            splitBf16(v.x, h.b[0], l.b[0]);
            splitBf16(v.y, h.b[1], l.b[1]);
            splitBf16(v.z, h.b[2], l.b[2]);
            splitBf16(v.w, h.b[3], l.b[3]);
            const int kk = hcolq * 4;
            *reinterpret_cast<u2*>(&Ahi[tok][kk]) = h.u;   // one ds_store_b64 each
            *reinterpret_cast<u2*>(&Alo[tok][kk]) = l.u;
        }
        // masked-mean pooling partials (dense head input) -> workspace
        float* pd = ws_pool + (size_t)b * H_ + kh * 512 + hcolq * 4;
        atomicAdd(pd + 0, pool.x); atomicAdd(pd + 1, pool.y);
        atomicAdd(pd + 2, pool.z); atomicAdd(pd + 3, pool.w);

        __syncthreads();

        // ---- bf16x3 WMMA over this K-half: each wave owns 8 contiguous N-tiles.
        // Hot loop: coalesced b128 loads + WMMA only (no conversion VALU).
        for (int kc = 0; kc < 16; ++kc) {
            const int kA = kc * 32 + hi16 * 16;  // A frag: one contiguous 16-elem run (matches B)
            Frag16 ah, al;
            ah.q[0] = *reinterpret_cast<const u4*>(&Ahi[l15][kA]);
            ah.q[1] = *reinterpret_cast<const u4*>(&Ahi[l15][kA + 8]);
            al.q[0] = *reinterpret_cast<const u4*>(&Alo[l15][kA]);
            al.q[1] = *reinterpret_cast<const u4*>(&Alo[l15][kA + 8]);
            const int c = kh * 16 + kc;          // global 32-wide K chunk index
#pragma unroll
            for (int j = 0; j < 8; ++j) {
                const int nt = wave * 8 + j;     // N tile index
                const size_t base = ((size_t)(nt * 32 + c) * 32 + lane) * 16;
                const bf16_t* wh = whi + base;   // wave reads one contiguous 1KB per plane
                const bf16_t* wl = wlo + base;
                Frag16 bh, bl;
                bh.q[0] = *reinterpret_cast<const u4*>(wh);
                bh.q[1] = *reinterpret_cast<const u4*>(wh + 8);
                bl.q[0] = *reinterpret_cast<const u4*>(wl);
                bl.q[1] = *reinterpret_cast<const u4*>(wl + 8);
                acc[j] = __builtin_amdgcn_wmma_f32_16x16x32_bf16(
                    false, ah.v, false, bh.v, (short)0, acc[j], false, false);
                acc[j] = __builtin_amdgcn_wmma_f32_16x16x32_bf16(
                    false, ah.v, false, bl.v, (short)0, acc[j], false, false);
                acc[j] = __builtin_amdgcn_wmma_f32_16x16x32_bf16(
                    false, al.v, false, bh.v, (short)0, acc[j], false, false);
            }
        }
        if (kh == 0) __syncthreads();   // before overwriting LDS with next K-half
    }

    // ---- sparse head: reduce token dots across wave then across waves, scatter
#pragma unroll
    for (int i = 0; i < 8; ++i) {
        float p = pA[i];
#pragma unroll
        for (int off = 16; off > 0; off >>= 1) p += __shfl_xor(p, off, 32);
        if (lane == 0) swp[wave][i] = p;
    }
    __syncthreads();
    if (wave == 0 && lane < 16) {
        const int tok = lane;
        const int ii = tok >> 1, pp = tok & 1;
        float s = 0.f;
#pragma unroll
        for (int w2 = 0; w2 < 4; ++w2) s += swp[pp * 4 + w2][ii];
        const float m  = mask[(size_t)b * S_ + s0 + tok];
        float tw = (s + sparse_b[0]) * m;
        tw = tw > 0.f ? tw : 0.f;
        const int id = input_ids[(size_t)b * S_ + s0 + tok];
        atomicAdd(out_sparse + (size_t)b * V_ + id, tw);
    }

    // ---- mv epilogue: bias, mask, per-token sum of squares (shfl + LDS atomics)
    float maskv[8];
#pragma unroll
    for (int r = 0; r < 8; ++r)
        maskv[r] = mask[(size_t)b * S_ + s0 + r + hi8];
#pragma unroll
    for (int j = 0; j < 8; ++j) {
        const int n = (wave * 8 + j) * 16 + l15;
        const float bias = mv_b[n];
        float ssq[8];
#pragma unroll
        for (int r = 0; r < 8; ++r) {
            float y = (acc[j][r] + bias) * maskv[r];
            acc[j][r] = y;
            ssq[r] = y * y;
        }
#pragma unroll
        for (int off = 1; off < 16; off <<= 1)
#pragma unroll
            for (int r = 0; r < 8; ++r) ssq[r] += __shfl_xor(ssq[r], off, 32);
        if (l15 == 0)
#pragma unroll
            for (int r = 0; r < 8; ++r) atomicAdd(&rowsq[r + hi8], ssq[r]);
    }
    __syncthreads();

    float inv[8];
#pragma unroll
    for (int r = 0; r < 8; ++r)
        inv[r] = 1.f / fmaxf(sqrtf(rowsq[r + hi8]), EPSN);
#pragma unroll
    for (int j = 0; j < 8; ++j) {
        const int n = (wave * 8 + j) * 16 + l15;
#pragma unroll
        for (int r = 0; r < 8; ++r)
            out_mv[(size_t)(t0 + r + hi8) * H_ + n] = acc[j][r] * inv[r];
    }
}

// ---------------------------------------------------------------------------
// Kernel 2: dense head — pooled/masksum -> linear -> tanh -> L2 norm (tiny)
// grid = 32 blocks (one per batch), 256 threads
// ---------------------------------------------------------------------------
__global__ __launch_bounds__(256)
void dense_head_kernel(const float* __restrict__ mask,
                       const float* __restrict__ dense_w,
                       const float* __restrict__ dense_b,
                       const float* __restrict__ ws_pool,
                       float* __restrict__ out_dense)
{
    __shared__ __align__(16) float spool[H_];
    __shared__ float red[256];
    __shared__ float sbc;
    const int b = blockIdx.x;
    const int t = threadIdx.x;

    red[t] = mask[(size_t)b * S_ + t] + mask[(size_t)b * S_ + 256 + t];
    __syncthreads();
    for (int s = 128; s > 0; s >>= 1) {
        if (t < s) red[t] += red[t + s];
        __syncthreads();
    }
    if (t == 0) sbc = 1.f / red[0];
    __syncthreads();
    const float invm = sbc;
#pragma unroll
    for (int j = 0; j < 4; ++j)
        spool[t + 256 * j] = ws_pool[(size_t)b * H_ + t + 256 * j] * invm;
    __syncthreads();

    float y[4]; float ss = 0.f;
#pragma unroll
    for (int j = 0; j < 4; ++j) {
        const int n = t + 256 * j;
        const f4* wr = reinterpret_cast<const f4*>(dense_w + (size_t)n * H_);
        const f4* sp = reinterpret_cast<const f4*>(spool);
        float dot = dense_b[n];
        for (int k = 0; k < H_ / 4; ++k) {
            const f4 w4 = wr[k], p4 = sp[k];
            dot += w4.x * p4.x + w4.y * p4.y + w4.z * p4.z + w4.w * p4.w;
        }
        y[j] = tanhf(dot);
        ss += y[j] * y[j];
    }
    __syncthreads();
    red[t] = ss;
    __syncthreads();
    for (int s = 128; s > 0; s >>= 1) {
        if (t < s) red[t] += red[t + s];
        __syncthreads();
    }
    if (t == 0) sbc = 1.f / fmaxf(sqrtf(red[0]), EPSN);
    __syncthreads();
    const float inv = sbc;
#pragma unroll
    for (int j = 0; j < 4; ++j)
        out_dense[(size_t)b * H_ + t + 256 * j] = y[j] * inv;
}

// ---------------------------------------------------------------------------
extern "C" void kernel_launch(void* const* d_in, const int* in_sizes, int n_in,
                              void* d_out, int out_size, void* d_ws, size_t ws_size,
                              hipStream_t stream)
{
    (void)in_sizes; (void)n_in; (void)out_size; (void)ws_size;

    const float* hidden   = (const float*)d_in[0];
    const int*   ids      = (const int*)d_in[1];   // per harness: integer -> const int*
    const float* mask     = (const float*)d_in[2];
    const float* dense_w  = (const float*)d_in[3];
    const float* dense_b  = (const float*)d_in[4];
    const float* sparse_w = (const float*)d_in[5];
    const float* sparse_b = (const float*)d_in[6];
    const float* mv_w     = (const float*)d_in[7];
    const float* mv_b     = (const float*)d_in[8];

    float* out_dense  = (float*)d_out;                       // [32,1024]
    float* out_sparse = out_dense + (size_t)B_ * H_;         // [32,250002]
    float* out_mv     = out_sparse + (size_t)B_ * V_;        // [32,512,1024]

    // workspace layout: pooled sums [32,1024] f32, then bf16 hi/lo weight planes
    float*  ws_pool = (float*)d_ws;
    bf16_t* whi     = (bf16_t*)((char*)d_ws + (size_t)B_ * H_ * sizeof(float));
    bf16_t* wlo     = whi + (size_t)H_ * H_;

    hipMemsetAsync(out_sparse, 0, (size_t)B_ * V_ * sizeof(float), stream);
    hipMemsetAsync(ws_pool,    0, (size_t)B_ * H_ * sizeof(float), stream);

    convert_w_kernel<<<(H_ * H_) / (256 * 4), 256, 0, stream>>>(mv_w, whi, wlo);

    mv_sparse_pool_kernel<<<(B_ * S_) / 16, 256, 0, stream>>>(
        hidden, ids, mask, sparse_w, sparse_b, whi, wlo, mv_b,
        out_sparse, out_mv, ws_pool);

    dense_head_kernel<<<B_, 256, 0, stream>>>(mask, dense_w, dense_b,
                                              ws_pool, out_dense);
}
